// LSSLLikeLayer_34153579937913
// MI455X (gfx1250) — matrix operand order repaired and checked by reading, not compiled, not run on hardware
//
#include <hip/hip_runtime.h>
#include <hip/hip_bf16.h>
#include <math.h>

// ---------------------------------------------------------------------------
// LSSL-like layer for MI455X (gfx1250, wave32, WMMA, TDM)
//   T=2048, BSZ=16, H=256, N=64, M=1
// Pipeline:
//   k0: tiled transpose x [T,B,H] f32 -> xT [H][B][T] f16 (channel slabs)
//   k0b: W f32 -> f16
//   k1: Krylov conv-kernel per channel: 63 matvecs + WMMA squarings (P and P^T
//       maintained so all LDS traffic is vectorized) + 31 WMMA block steps;
//       emits kernel reversed + zero-padded for the Toeplitz stage
//   k2: causal Toeplitz conv per channel via WMMA (TDM stages slabs to LDS),
//       + D*x, exact GELU -> h [T*B][H] f16
//   k3: output GEMM h @ W^T + b via WMMA (TDM stages h tile) -> out f32
// ---------------------------------------------------------------------------

#define LT    2048
#define LB    16
#define LH    256
#define LN    64
#define LDK   72           // padded f16 leading dim for 64x64 LDS matrices
#define KREV  2080         // 2048 + 32 zero pad (reversed kernel)

typedef __attribute__((ext_vector_type(16))) _Float16 v16h;
typedef __attribute__((ext_vector_type(8)))  _Float16 v8h;
typedef __attribute__((ext_vector_type(8)))  float    v8f;
typedef __attribute__((ext_vector_type(4)))  unsigned u32x4;
typedef __attribute__((ext_vector_type(8)))  int      i32x8;
typedef __attribute__((ext_vector_type(4)))  int      i32x4;

#if defined(__has_builtin)
#if __has_builtin(__builtin_amdgcn_tensor_load_to_lds) && __has_builtin(__builtin_amdgcn_s_wait_tensorcnt)
#define LSSL_USE_TDM 1
#endif
#endif

// ----------------------------- WMMA helpers --------------------------------
static __device__ inline v16h lssl_cat(v8h lo, v8h hi) {
  return __builtin_shufflevector(lo, hi, 0, 1, 2, 3, 4, 5, 6, 7,
                                 8, 9, 10, 11, 12, 13, 14, 15);
}

// A fragment (16x32 MxK) from row-major S; per-lane data is one contiguous
// 16xf16 run: [k0+half*16 .. +15] of row m0+m.
static __device__ inline v16h lssl_load_a(const _Float16* S, int m0, int k0, int ld) {
  const int lane = threadIdx.x & 31;
  const int half = lane >> 4;
  const int m    = lane & 15;
  const _Float16* p = &S[(m0 + m) * ld + k0 + half * 16];
  return lssl_cat(*(const v8h*)p, *(const v8h*)(p + 8));
}

// B fragment (32x16 KxN) read from the TRANSPOSED matrix ST (N-major rows):
// lane covers column n, k = e + half*16 -> one contiguous 16xf16 run.
static __device__ inline v16h lssl_load_bT(const _Float16* ST, int k0, int n0, int ld) {
  const int lane = threadIdx.x & 31;
  const int half = lane >> 4;
  const int n    = lane & 15;
  const _Float16* p = &ST[(n0 + n) * ld + k0 + half * 16];
  return lssl_cat(*(const v8h*)p, *(const v8h*)(p + 8));
}

// Store O^T for a C/D fragment of O: per lane the 8 rows (r) map to 8
// consecutive columns of O^T -> a single 16-byte store.
static __device__ inline void lssl_store_cdT(_Float16* ST, int m0, int n0, int ld, v8f c) {
  const int lane = threadIdx.x & 31;
  const int half = lane >> 4;
  const int n    = lane & 15;
  *(v8h*)&ST[(n0 + n) * ld + m0 + half * 8] = __builtin_convertvector(c, v8h);
}

static __device__ inline v8f lssl_wmma(v16h a, v16h b, v8f c) {
  return __builtin_amdgcn_wmma_f32_16x16x32_f16(false, a, false, b, (short)0, c,
                                                false, false);
}

// ------------------------------ TDM helper ---------------------------------
#if defined(LSSL_USE_TDM)
// Contiguous 1-D slab load via Tensor Data Mover (8-byte elements so the
// 16-bit tile_dim0 covers up to 512KB). nbytes must be a multiple of 8.
static __device__ inline void lssl_tdm_load_linear(unsigned lds_addr, const void* src,
                                                   unsigned nbytes) {
  const unsigned long long ga = (unsigned long long)(uintptr_t)src;
  const unsigned n8 = nbytes >> 3;

  u32x4 g0;
  g0[0] = 1u;                                   // count=1
  g0[1] = lds_addr;                             // lds_addr
  g0[2] = (unsigned)(ga & 0xffffffffu);         // global_addr lo
  g0[3] = (unsigned)((ga >> 32) & 0x1ffffffu)   // global_addr hi
        | (2u << 30);                           // type=2

  i32x8 g1;
  g1[0] = (int)(3u << 16);                      // data_size=3 (8B)
  g1[1] = (int)((n8 & 0xffffu) << 16);          // tensor_dim0 lo16
  g1[2] = (int)(((n8 >> 16) & 0xffffu) | (1u << 16));  // dim0 hi16 | dim1=1
  g1[3] = (int)((n8 & 0xffffu) << 16);          // tile_dim0
  g1[4] = 1;                                    // tile_dim1=1
  g1[5] = (int)n8;                              // dim0 stride
  g1[6] = 0;
  g1[7] = 0;

  i32x4 z4 = {0, 0, 0, 0};
  i32x8 z8 = {0, 0, 0, 0, 0, 0, 0, 0};
  __builtin_amdgcn_tensor_load_to_lds(g0, g1, z4, z4, z8, 0);
}
#endif

// ----------------------- k0: tiled transpose of x --------------------------
// xT[c][b][t] = (f16) x[t][b][c], via 64x64 LDS tiles (odd-dword row pad).
__global__ __launch_bounds__(256) void lssl_transpose_kernel(const float* __restrict__ x,
                                                             _Float16* __restrict__ xTh) {
  __shared__ _Float16 tile[64 * 66];           // 66 f16 = 33 dwords (odd)

  const int bid = blockIdx.x;                  // 128 blocks
  const int t0  = (bid & 31) * 64;
  const int c0  = (bid >> 5) * 64;
  const int tid = threadIdx.x;
  const int jj  = tid & 63;                    // inner (c on read, t on write)
  const int i0  = tid >> 6;                    // 0..3

  for (int b = 0; b < LB; ++b) {
    __syncthreads();
#pragma unroll
    for (int ii = 0; ii < 64; ii += 4) {
      const int t = t0 + ii + i0;
      tile[(ii + i0) * 66 + jj] = (_Float16)x[(size_t)t * (LB * LH) + b * LH + c0 + jj];
    }
    __syncthreads();
#pragma unroll
    for (int ii = 0; ii < 64; ii += 4) {
      const int c = c0 + ii + i0;
      xTh[(size_t)c * (LB * LT) + b * LT + t0 + jj] = tile[jj * 66 + (ii + i0)];
    }
  }
}

// ------------------------- k0b: convert W to f16 ---------------------------
__global__ __launch_bounds__(256) void lssl_wcvt_kernel(const float* __restrict__ W,
                                                        _Float16* __restrict__ Wh) {
  const int i = blockIdx.x * 256 + threadIdx.x;
  Wh[i] = (_Float16)W[i];
}

// ------------------------ k1: Krylov kernel (per channel) ------------------
// Static __shared__ double buffers indexed directly (PhS[pc]) keep every
// access in addrspace(3) -> ds_load_b128/ds_store_b128 (no flat fallback).
__global__ __launch_bounds__(64) void lssl_krylov_kernel(const float* __restrict__ A,
                                                         const float* __restrict__ Bv,
                                                         const float* __restrict__ C,
                                                         _Float16* __restrict__ krevh) {
  __shared__ _Float16 PhS[2][LN * LDK];        // P   (row-major), 18432 B
  __shared__ _Float16 PhTS[2][LN * LDK];       // P^T (row-major), 18432 B
  __shared__ _Float16 KTS[2][LN * LDK];        // K^T: [t][n],     18432 B
  __shared__ float    vbuf[2][LN];
  __shared__ float    Cc[LN];

  const int c    = blockIdx.x;
  const int tid  = threadIdx.x;              // 64 threads = 2 waves
  const int wave = tid >> 5;

  for (int idx = tid; idx < LN * LN; idx += 64) {
    const int r = idx >> 6, cl = idx & 63;
    const _Float16 h = (_Float16)A[(size_t)c * LN * LN + idx];
    PhS[0][r * LDK + cl]  = h;   // P = A
    PhTS[0][cl * LDK + r] = h;   // P^T
  }
  Cc[tid]      = C[c * LN + tid];
  vbuf[0][tid] = Bv[c * LN + tid];
  if (tid < 32) krevh[(size_t)c * KREV + LT + tid] = (_Float16)0.f;  // zero pad
  __syncthreads();

  // --- first 64 columns via sequential matvec chain (stored as K^T) -------
  KTS[0][0 * LDK + tid] = (_Float16)vbuf[0][tid];
  int cur = 0;
  for (int t = 1; t < LN; ++t) {
    float s = 0.f;
    const _Float16* row = &PhS[0][tid * LDK];
#pragma unroll 8
    for (int j = 0; j < LN; ++j) s += (float)row[j] * vbuf[cur][j];
    vbuf[1 - cur][tid]   = s;
    KTS[0][t * LDK + tid] = (_Float16)s;
    cur ^= 1;
    __syncthreads();
  }

  // kern (reversed) for block 0: thread tid owns column t = tid
  {
    float s = 0.f;
#pragma unroll 8
    for (int n = 0; n < LN; ++n) s += Cc[n] * (float)KTS[0][tid * LDK + n];
    krevh[(size_t)c * KREV + (LT - 1 - tid)] = (_Float16)s;
  }
  __syncthreads();

  // --- A64 = A^64: 6 squarings, maintaining P and P^T ----------------------
  // O1 = P@P   (A rows of P,  B cols of P  = rows of P^T) -> T-store = Pn^T
  // O2 = Pt@Pt (A rows of P^T, B cols of P^T = rows of P) -> T-store = Pn
  int pc = 0;
  for (int it = 0; it < 6; ++it) {
    for (int tile = wave; tile < 16; tile += 2) {
      const int mt = (tile >> 2) * 16, nt = (tile & 3) * 16;
      v8f acc1 = {}, acc2 = {};
#pragma unroll
      for (int kb = 0; kb < 2; ++kb) {
        acc1 = lssl_wmma(lssl_load_a(PhS[pc], mt, kb * 32, LDK),
                         lssl_load_bT(PhTS[pc], kb * 32, nt, LDK), acc1);
        acc2 = lssl_wmma(lssl_load_a(PhTS[pc], mt, kb * 32, LDK),
                         lssl_load_bT(PhS[pc], kb * 32, nt, LDK), acc2);
      }
      lssl_store_cdT(PhTS[1 - pc], mt, nt, LDK, acc1);
      lssl_store_cdT(PhS[1 - pc],  mt, nt, LDK, acc2);
    }
    __syncthreads();
    pc ^= 1;
  }

  // --- K_{j+1} = A64 @ K_j (T-stored), then dot with C per column ----------
  int kc = 0;
  for (int j = 1; j < 32; ++j) {
    for (int tile = wave; tile < 16; tile += 2) {
      const int mt = (tile >> 2) * 16, nt = (tile & 3) * 16;
      v8f acc = {};
#pragma unroll
      for (int kb = 0; kb < 2; ++kb) {
        acc = lssl_wmma(lssl_load_a(PhS[pc], mt, kb * 32, LDK),
                        lssl_load_bT(KTS[kc], kb * 32, nt, LDK), acc);
      }
      lssl_store_cdT(KTS[1 - kc], mt, nt, LDK, acc);
    }
    __syncthreads();
    kc ^= 1;
    float s = 0.f;
#pragma unroll 8
    for (int n = 0; n < LN; ++n) s += Cc[n] * (float)KTS[kc][tid * LDK + n];
    krevh[(size_t)c * KREV + (LT - 1 - (j * LN + tid))] = (_Float16)s;
    __syncthreads();
  }
}

// -------------------- k2: causal Toeplitz conv (per channel) ---------------
// y[t,b] = sum_{s<=t} kern[t-s] * x[s,b]; Toeplitz A-frag comes from the
// reversed zero-padded kernel so all taps are plain ascending vector loads.
__global__ __launch_bounds__(256) void lssl_conv_kernel(const _Float16* __restrict__ xTh,
                                                        const _Float16* __restrict__ krevh,
                                                        const float* __restrict__ Dv,
                                                        _Float16* __restrict__ hh) {
  extern __shared__ char smem[];
  _Float16* xs   = (_Float16*)smem;                    // [B][T] = 64 KB
  _Float16* krev = (_Float16*)(smem + LT * LB * 2);    // [KREV] = 4160 B

  const int c    = blockIdx.x;
  const int tid  = threadIdx.x;               // 256 threads = 8 waves
  const int wave = tid >> 5;
  const int lane = tid & 31;

#if defined(LSSL_USE_TDM)
  if (wave == 0) {
    lssl_tdm_load_linear((unsigned)(uintptr_t)xs, xTh + (size_t)c * LT * LB,
                         LT * LB * sizeof(_Float16));
    lssl_tdm_load_linear((unsigned)(uintptr_t)krev, krevh + (size_t)c * KREV,
                         KREV * sizeof(_Float16));
    __builtin_amdgcn_s_wait_tensorcnt(0);
  }
#else
  for (int i = tid; i < LT * LB; i += 256) xs[i]   = xTh[(size_t)c * LT * LB + i];
  for (int i = tid; i < KREV;    i += 256) krev[i] = krevh[(size_t)c * KREV + i];
#endif
  __syncthreads();

  const float Dc   = Dv[c];
  const int   half = lane >> 4;
  const int   n    = lane & 15;
  const int   m    = lane & 15;

  for (int ti = wave; ti < LT / 16; ti += 8) {
    v8f acc0 = {}, acc1 = {};
    const int kbmax = (16 * ti + 15) >> 5;    // inclusive
    int kb = 0;
    for (; kb + 1 <= kbmax; kb += 2) {
      {
        const _Float16* p = &krev[(LT - 1) - 16 * ti - m + 32 * kb + half * 16];
        v16h af = lssl_cat(*(const v8h*)p, *(const v8h*)(p + 8));
        const _Float16* q = &xs[n * LT + 32 * kb + half * 16];
        v16h bf = lssl_cat(*(const v8h*)q, *(const v8h*)(q + 8));
        acc0 = lssl_wmma(af, bf, acc0);
      }
      {
        const _Float16* p = &krev[(LT - 1) - 16 * ti - m + 32 * (kb + 1) + half * 16];
        v16h af = lssl_cat(*(const v8h*)p, *(const v8h*)(p + 8));
        const _Float16* q = &xs[n * LT + 32 * (kb + 1) + half * 16];
        v16h bf = lssl_cat(*(const v8h*)q, *(const v8h*)(q + 8));
        acc1 = lssl_wmma(af, bf, acc1);
      }
    }
    if (kb <= kbmax) {
      const _Float16* p = &krev[(LT - 1) - 16 * ti - m + 32 * kb + half * 16];
      v16h af = lssl_cat(*(const v8h*)p, *(const v8h*)(p + 8));
      const _Float16* q = &xs[n * LT + 32 * kb + half * 16];
      v16h bf = lssl_cat(*(const v8h*)q, *(const v8h*)(q + 8));
      acc0 = lssl_wmma(af, bf, acc0);
    }
    v8f acc = acc0 + acc1;

    // epilogue: +D*x, exact GELU, scatter h[(t*B+b), c] f16
    const v8h xv8 = *(const v8h*)&xs[n * LT + 16 * ti + half * 8];
#pragma unroll
    for (int r = 0; r < 8; ++r) {
      const int t  = 16 * ti + half * 8 + r;
      const float y = acc[r] + Dc * (float)xv8[r];
      const float g = 0.5f * y * (1.0f + erff(y * 0.70710678118654752f));
      hh[(size_t)(t * LB + n) * LH + c] = (_Float16)g;
    }
  }
}

// ----------------------- k3: output GEMM h @ W^T + b -----------------------
__global__ __launch_bounds__(256) void lssl_out_gemm_kernel(const _Float16* __restrict__ hh,
                                                            const _Float16* __restrict__ Wh,
                                                            const float* __restrict__ bias,
                                                            float* __restrict__ out) {
  __shared__ _Float16 hs[16 * LH];            // one M-tile of h rows, 8 KB

  const int mt   = blockIdx.x;                // 2048 row tiles
  const int tid  = threadIdx.x;               // 8 waves
  const int wave = tid >> 5;
  const int lane = tid & 31;
  const int half = lane >> 4;
  const int n    = lane & 15;

#if defined(LSSL_USE_TDM)
  if (wave == 0) {
    lssl_tdm_load_linear((unsigned)(uintptr_t)hs, hh + (size_t)mt * 16 * LH,
                         16 * LH * sizeof(_Float16));
    __builtin_amdgcn_s_wait_tensorcnt(0);
  }
#else
  for (int i = tid; i < 16 * LH; i += 256) hs[i] = hh[(size_t)mt * 16 * LH + i];
#endif
  __syncthreads();

  for (int nt = wave; nt < 16; nt += 8) {
    v8f acc0 = {}, acc1 = {};
#pragma unroll
    for (int kb = 0; kb < 8; kb += 2) {
      {
        v16h af = lssl_load_a(hs, 0, kb * 32, LH);
        const _Float16* q = &Wh[(size_t)(16 * nt + n) * LH + kb * 32 + half * 16];
        v16h bf = lssl_cat(*(const v8h*)q, *(const v8h*)(q + 8));
        acc0 = lssl_wmma(af, bf, acc0);
      }
      {
        v16h af = lssl_load_a(hs, 0, (kb + 1) * 32, LH);
        const _Float16* q = &Wh[(size_t)(16 * nt + n) * LH + (kb + 1) * 32 + half * 16];
        v16h bf = lssl_cat(*(const v8h*)q, *(const v8h*)(q + 8));
        acc1 = lssl_wmma(af, bf, acc1);
      }
    }
    v8f acc = acc0 + acc1;
#pragma unroll
    for (int r = 0; r < 8; ++r) {
      const int row = mt * 16 + half * 8 + r;
      const int o   = 16 * nt + n;
      out[(size_t)row * LH + o] = acc[r] + bias[o];
    }
  }
}

// ------------------------------- launcher ----------------------------------
extern "C" void kernel_launch(void* const* d_in, const int* in_sizes, int n_in,
                              void* d_out, int out_size, void* d_ws, size_t ws_size,
                              hipStream_t stream) {
  (void)in_sizes; (void)n_in; (void)out_size; (void)ws_size;

  const float* x    = (const float*)d_in[0];  // [T, B, H]
  const float* A    = (const float*)d_in[1];  // [H, N, N]
  const float* Bv   = (const float*)d_in[2];  // [H, N]
  const float* C    = (const float*)d_in[3];  // [H, 1, N]
  const float* D    = (const float*)d_in[4];  // [H, 1]
  const float* W    = (const float*)d_in[5];  // [H, H]
  const float* bias = (const float*)d_in[6];  // [H]
  float*       out  = (float*)d_out;          // [T, B, H]

  char* ws = (char*)d_ws;
  _Float16* xTh   = (_Float16*)(ws);                                  // 16 MB
  _Float16* krevh = (_Float16*)(ws + (size_t)16 * 1024 * 1024);       // ~1 MB
  _Float16* hh    = (_Float16*)(ws + (size_t)18 * 1024 * 1024);       // 16 MB
  _Float16* Wh    = (_Float16*)(ws + (size_t)34 * 1024 * 1024);       // 128 KB

  lssl_transpose_kernel<<<(LT / 64) * (LH / 64), 256, 0, stream>>>(x, xTh);
  lssl_wcvt_kernel<<<(LH * LH) / 256, 256, 0, stream>>>(W, Wh);
  lssl_krylov_kernel<<<LH, 64, 0, stream>>>(A, Bv, C, krevh);
  lssl_conv_kernel<<<LH, 256, LT * LB * 2 + KREV * 2, stream>>>(xTh, krevh, D, hh);
  lssl_out_gemm_kernel<<<(LT * LB) / 16, 256, 0, stream>>>(hh, Wh, bias, out);
}
